// EFR_51153060496132
// MI455X (gfx1250) — matrix operand order repaired and checked by reading, not compiled, not run on hardware
//
#include <hip/hip_runtime.h>

typedef __attribute__((ext_vector_type(16))) _Float16 v16h;
typedef __attribute__((ext_vector_type(8)))  float    v8f;

#define BATCH 2
#define CH    64
#define HH    192
#define WW    192
#define HW    (HH*WW)
#define EPSF  1e-5f

// ---------------- Stage A: fused depthwise convs (1x15 + 15x1 + 3x3) + BN sum ----------------
__global__ __launch_bounds__(256) void offsum_kernel(
    const float* __restrict__ x,
    const float* __restrict__ w1, const float* __restrict__ b1, const float* __restrict__ bn1,
    const float* __restrict__ w2, const float* __restrict__ b2, const float* __restrict__ bn2,
    const float* __restrict__ w3, const float* __restrict__ b3, const float* __restrict__ bn3,
    float* __restrict__ off_sum)
{
    int idx  = blockIdx.x * 256 + threadIdx.x;
    int wcol = idx % WW;
    int hrow = (idx / WW) % HH;
    int c    = (idx / HW) % CH;
    int b    = idx / (HW * CH);
    const float* xp = x + (size_t)(b * CH + c) * HW;

    float s1 = 0.f;
    #pragma unroll
    for (int k = 0; k < 15; ++k) {
        int cc = wcol + k - 7;
        float v = (cc >= 0 && cc < WW) ? xp[hrow * WW + cc] : 0.f;
        s1 += v * w1[c * 15 + k];
    }
    float s2 = 0.f;
    #pragma unroll
    for (int k = 0; k < 15; ++k) {
        int rr = hrow + k - 7;
        float v = (rr >= 0 && rr < HH) ? xp[rr * WW + wcol] : 0.f;
        s2 += v * w2[c * 15 + k];
    }
    float s3 = 0.f;
    #pragma unroll
    for (int ky = 0; ky < 3; ++ky)
        #pragma unroll
        for (int kx = 0; kx < 3; ++kx) {
            int rr = hrow + ky - 1, cc = wcol + kx - 1;
            float v = (rr >= 0 && rr < HH && cc >= 0 && cc < WW) ? xp[rr * WW + cc] : 0.f;
            s3 += v * w3[c * 9 + ky * 3 + kx];
        }
    float a1 = bn1[c] * rsqrtf(bn1[3 * CH + c] + EPSF);
    float a2 = bn2[c] * rsqrtf(bn2[3 * CH + c] + EPSF);
    float a3 = bn3[c] * rsqrtf(bn3[3 * CH + c] + EPSF);
    float r = (s1 + b1[c] - bn1[2 * CH + c]) * a1 + bn1[CH + c]
            + (s2 + b2[c] - bn2[2 * CH + c]) * a2 + bn2[CH + c]
            + (s3 + b3[c] - bn3[2 * CH + c]) * a3 + bn3[CH + c];
    off_sum[idx] = r;
}

// ---------------- Stage B: offsets = bn(1x1 conv 64->18) via WMMA ----------------
__global__ __launch_bounds__(256) void offsets_gemm_kernel(
    const float* __restrict__ off_sum,
    const float* __restrict__ w,      // [18][64]
    const float* __restrict__ bias,   // [18]
    const float* __restrict__ bnp,    // [4][18]
    float* __restrict__ offsets)      // [B][18][HW]
{
    int lane = threadIdx.x & 31;
    int wave = threadIdx.x >> 5;      // 0..7
    int coutTile = wave & 1;          // 0..1 (covers cout 0..31, only 0..17 valid)
    int pxSub = wave >> 1;            // 0..3
    int b  = blockIdx.y;
    int p0 = (blockIdx.x * 4 + pxSub) * 16;
    int g  = lane >> 4;
    int ln = lane & 15;
    int M  = coutTile * 16 + ln;
    int p  = p0 + ln;

    v16h a0, a1;
    float msk = (M < 18) ? 1.f : 0.f;
    int Msafe = (M < 18) ? M : 0;
    #pragma unroll
    for (int j = 0; j < 16; ++j) {
        int k = ((j < 8) ? j : j + 8) + 8 * g;
        a0[j] = (_Float16)(w[Msafe * 64 + k] * msk);
        a1[j] = (_Float16)(w[Msafe * 64 + k + 32] * msk);
    }
    const float* src = off_sum + (size_t)b * CH * HW + p;
    v16h bm0, bm1;
    #pragma unroll
    for (int j = 0; j < 16; ++j) {
        int cin = ((j < 8) ? j : j + 8) + 8 * g;
        bm0[j] = (_Float16)src[(size_t)cin * HW];
        bm1[j] = (_Float16)src[(size_t)(cin + 32) * HW];
    }
    v8f c = {};
    c = __builtin_amdgcn_wmma_f32_16x16x32_f16(false, a0, false, bm0, (short)0, c, false, false);
    c = __builtin_amdgcn_wmma_f32_16x16x32_f16(false, a1, false, bm1, (short)0, c, false, false);

    #pragma unroll
    for (int r = 0; r < 8; ++r) {
        int cout = coutTile * 16 + r + 8 * g;
        if (cout < 18) {
            float s   = bnp[cout] * rsqrtf(bnp[3 * 18 + cout] + EPSF);
            float val = c[r] * s + (bias[cout] - bnp[2 * 18 + cout]) * s + bnp[18 + cout];
            offsets[(size_t)b * 18 * HW + (size_t)cout * HW + p] = val;
        }
    }
}

// ---------------- Stage C: deformable sampling with LDS-shared sample descriptors ----------
// Block = 32 pixels x 64 channels. Phase 1: compute 27x32 sample descriptors
// (4 clamped plane offsets + 4 validity-folded bilinear weights) ONCE into LDS.
// Phase 2: each thread accumulates 8 channels for its pixel via ds_load_b128 + gathers.
#define NPIX 32
#define NENT (27 * NPIX)   // 864 descriptors per block

__global__ __launch_bounds__(256) void deform_kernel(
    const float* __restrict__ x,
    const float* __restrict__ offsets, // [B][18][HW]
    const float* __restrict__ wdef,    // [3][C][3][3]
    float* __restrict__ deform)        // [B][C][HW]
{
    __shared__ int4   sOff[NENT];      // clamped plane offsets (4 corners)
    __shared__ float4 sWt[NENT];       // bilinear weights * validity

    int b  = blockIdx.y;
    int p0 = blockIdx.x * NPIX;
    const float* ob = offsets + (size_t)b * 18 * HW;

    // ---- Phase 1: 864 descriptors, cooperatively ----
    for (int e = threadIdx.x; e < NENT; e += 256) {
        int t  = e / (3 * NPIX);       // tap 0..8
        int i  = (e / NPIX) % 3;       // dilation index
        int pi = e % NPIX;
        int p  = p0 + pi;
        int hrow = p / WW, wcol = p % WW;
        int dil = 1 << i;              // 1,2,4
        float offy = ob[(size_t)(2 * t) * HW + p];
        float offx = ob[(size_t)(2 * t + 1) * HW + p];
        float yy = offy + (float)hrow + (float)((t / 3 - 1) * dil);
        float xx = offx + (float)wcol + (float)((t % 3 - 1) * dil);
        float y0f = floorf(yy), x0f = floorf(xx);
        float wy1 = yy - y0f, wx1 = xx - x0f;
        float wy0 = 1.f - wy1, wx0 = 1.f - wx1;
        int y0 = (int)y0f, x0 = (int)x0f;
        int y1 = y0 + 1,   x1 = x0 + 1;
        float vy0 = (y0 >= 0 && y0 < HH) ? 1.f : 0.f;
        float vy1 = (y1 >= 0 && y1 < HH) ? 1.f : 0.f;
        float vx0 = (x0 >= 0 && x0 < WW) ? 1.f : 0.f;
        float vx1 = (x1 >= 0 && x1 < WW) ? 1.f : 0.f;
        int y0c = min(max(y0, 0), HH - 1), y1c = min(max(y1, 0), HH - 1);
        int x0c = min(max(x0, 0), WW - 1), x1c = min(max(x1, 0), WW - 1);
        sOff[e] = make_int4(y0c * WW + x0c, y0c * WW + x1c,
                            y1c * WW + x0c, y1c * WW + x1c);
        sWt[e]  = make_float4(wy0 * wx0 * vy0 * vx0, wy0 * wx1 * vy0 * vx1,
                              wy1 * wx0 * vy1 * vx0, wy1 * wx1 * vy1 * vx1);
    }
    __syncthreads();

    // ---- Phase 2: 8 channels per thread ----
    int pi = threadIdx.x & 31;         // lane = pixel -> coalesced gathers per wave
    int cg = threadIdx.x >> 5;         // channel group 0..7
    int p  = p0 + pi;
    const float* xb = x + (size_t)b * CH * HW;

    float acc[8] = {0.f, 0.f, 0.f, 0.f, 0.f, 0.f, 0.f, 0.f};
    for (int t = 0; t < 9; ++t) {
        for (int i = 0; i < 3; ++i) {
            int e = (t * 3 + i) * NPIX + pi;
            int4   o  = sOff[e];
            float4 wt = sWt[e];
            #pragma unroll
            for (int c8 = 0; c8 < 8; ++c8) {
                int c = cg * 8 + c8;
                const float* plane = xb + (size_t)c * HW;
                float sv = plane[o.x] * wt.x + plane[o.y] * wt.y
                         + plane[o.z] * wt.z + plane[o.w] * wt.w;
                acc[c8] += sv * wdef[(size_t)(i * CH + c) * 9 + t];
            }
        }
    }
    #pragma unroll
    for (int c8 = 0; c8 < 8; ++c8) {
        int c = cg * 8 + c8;
        deform[(size_t)b * CH * HW + (size_t)c * HW + p] = acc[c8];
    }
}

// ---------------- Stage D: out = bn(1x1 conv 64->64 of deform) * x via WMMA ----------------
__global__ __launch_bounds__(256) void final_gemm_kernel(
    const float* __restrict__ deform,
    const float* __restrict__ x,
    const float* __restrict__ w,     // [64][64]
    const float* __restrict__ bias,  // [64]
    const float* __restrict__ bnp,   // [4][64]
    float* __restrict__ out)
{
    int lane = threadIdx.x & 31;
    int wave = threadIdx.x >> 5;
    int coutTile = wave & 3;
    int pxSub = wave >> 2;
    int b  = blockIdx.y;
    int p0 = (blockIdx.x * 2 + pxSub) * 16;
    int g  = lane >> 4;
    int ln = lane & 15;
    int M  = coutTile * 16 + ln;
    int p  = p0 + ln;

    // prefetch the x rows the epilogue multiplies against (overlaps with WMMA)
    __builtin_prefetch(x + (size_t)b * CH * HW + (size_t)(coutTile * 16 + 8 * g) * HW + p, 0, 3);

    v16h a0, a1;
    #pragma unroll
    for (int j = 0; j < 16; ++j) {
        int k = ((j < 8) ? j : j + 8) + 8 * g;
        a0[j] = (_Float16)w[M * 64 + k];
        a1[j] = (_Float16)w[M * 64 + k + 32];
    }
    const float* src = deform + (size_t)b * CH * HW + p;
    v16h bm0, bm1;
    #pragma unroll
    for (int j = 0; j < 16; ++j) {
        int cin = ((j < 8) ? j : j + 8) + 8 * g;
        bm0[j] = (_Float16)src[(size_t)cin * HW];
        bm1[j] = (_Float16)src[(size_t)(cin + 32) * HW];
    }
    v8f c = {};
    c = __builtin_amdgcn_wmma_f32_16x16x32_f16(false, a0, false, bm0, (short)0, c, false, false);
    c = __builtin_amdgcn_wmma_f32_16x16x32_f16(false, a1, false, bm1, (short)0, c, false, false);

    #pragma unroll
    for (int r = 0; r < 8; ++r) {
        int cout = coutTile * 16 + r + 8 * g;
        float s   = bnp[cout] * rsqrtf(bnp[3 * CH + cout] + EPSF);
        float val = c[r] * s + (bias[cout] - bnp[2 * CH + cout]) * s + bnp[CH + cout];
        size_t oidx = (size_t)b * CH * HW + (size_t)cout * HW + p;
        out[oidx] = val * x[oidx];
    }
}

extern "C" void kernel_launch(void* const* d_in, const int* in_sizes, int n_in,
                              void* d_out, int out_size, void* d_ws, size_t ws_size,
                              hipStream_t stream)
{
    (void)in_sizes; (void)n_in; (void)out_size; (void)ws_size;
    const float* x          = (const float*)d_in[0];
    const float* w_off1     = (const float*)d_in[1];
    const float* b_off1     = (const float*)d_in[2];
    const float* bn_off1    = (const float*)d_in[3];
    const float* w_off2     = (const float*)d_in[4];
    const float* b_off2     = (const float*)d_in[5];
    const float* bn_off2    = (const float*)d_in[6];
    const float* w_off3     = (const float*)d_in[7];
    const float* b_off3     = (const float*)d_in[8];
    const float* bn_off3    = (const float*)d_in[9];
    const float* w_bal_off  = (const float*)d_in[10];
    const float* b_bal_off  = (const float*)d_in[11];
    const float* bn_bal_off = (const float*)d_in[12];
    const float* w_def      = (const float*)d_in[13];
    const float* w_bal      = (const float*)d_in[14];
    const float* b_bal      = (const float*)d_in[15];
    const float* bn_bal     = (const float*)d_in[16];
    float* out = (float*)d_out;

    float* off_sum = (float*)d_ws;                          // B*C*HW floats (18.9 MB), reused
    float* offsets = off_sum + (size_t)BATCH * CH * HW;     // B*18*HW floats (5.3 MB)

    const int total = BATCH * CH * HW;

    offsum_kernel<<<total / 256, 256, 0, stream>>>(
        x, w_off1, b_off1, bn_off1, w_off2, b_off2, bn_off2,
        w_off3, b_off3, bn_off3, off_sum);

    dim3 gB(HW / 64, BATCH);
    offsets_gemm_kernel<<<gB, 256, 0, stream>>>(
        off_sum, w_bal_off, b_bal_off, bn_bal_off, offsets);

    dim3 gC(HW / NPIX, BATCH);   // 1152 x 2 blocks, 32 pixels x 64 channels each
    deform_kernel<<<gC, 256, 0, stream>>>(
        x, offsets, w_def, off_sum);   // overwrites off_sum with deform_sum

    dim3 gD(HW / 32, BATCH);
    final_gemm_kernel<<<gD, 256, 0, stream>>>(
        off_sum, x, w_bal, b_bal, bn_bal, out);
}
